// CategoricalDecoder_58067957842650
// MI455X (gfx1250) — matrix operand order repaired and checked by reading, not compiled, not run on hardware
//
#include <hip/hip_runtime.h>
#include <hip/hip_bf16.h>
#include <math.h>

// Problem constants (from the reference)
#define BATCH 256
#define NBINS 8192
#define LDIM  64
#define HDIM  256
#define DFEAT 32
#define CCAT  16
#define DC    512           // DFEAT * CCAT
#define KTOP  16
#define YNUM  4

typedef __attribute__((ext_vector_type(16))) _Float16 v16h;
typedef __attribute__((ext_vector_type(8)))  _Float16 v8h;
typedef __attribute__((ext_vector_type(8)))  float    v8f;

#define ZS_STRIDE  72       // padded (halves) to avoid LDS bank conflicts; 144B = 9*16B
#define HID_STRIDE 264      // 528B = 33*16B -> v8h loads stay 16B-aligned

// -------------------------------------------------------------------------
// Kernel 0: one-time weight prep. Transpose W1 (64x256) and W2 (256x512) to
// column-major f16 so WMMA B-fragments become contiguous 32B vector loads.
//   W1T[c*64  + k] = (f16) W1[k*256 + c]
//   W2T[c*256 + k] = (f16) W2[k*512 + c]
// -------------------------------------------------------------------------
__global__ __launch_bounds__(256) void prep_kernel(
    const float* __restrict__ W1, const float* __restrict__ W2,
    _Float16* __restrict__ W1T, _Float16* __restrict__ W2T)
{
  const int i = blockIdx.x * 256 + threadIdx.x;
  if (i < LDIM * HDIM) {
    const int c = i / LDIM, k = i % LDIM;
    W1T[i] = (_Float16)W1[k * HDIM + c];
  }
  const int j = i - LDIM * HDIM;
  if (j >= 0 && j < HDIM * DC) {
    const int c = j / HDIM, k = j % HDIM;
    W2T[j] = (_Float16)W2[k * DC + c];
  }
}

// -------------------------------------------------------------------------
// Kernel 1: fused MLP (z@W1+b1 -> relu -> @W2+b2) + log_softmax over C=16.
// One workgroup = 16 bins; 256 threads = 8 wave32s; WMMA f16->f32.
// -------------------------------------------------------------------------
__global__ __launch_bounds__(256) void mlp_logp_kernel(
    const float* __restrict__ z,   const _Float16* __restrict__ W1T,
    const float* __restrict__ b1,  const _Float16* __restrict__ W2T,
    const float* __restrict__ b2,  float* __restrict__ logp)
{
  __shared__ _Float16 zs[16 * ZS_STRIDE];     // 16x64 z tile (f16)
  __shared__ _Float16 hid[16 * HID_STRIDE];   // 16x256 hidden tile (f16)

  const int tid  = threadIdx.x;
  const int row0 = blockIdx.x * 16;           // first bin of this tile
  const int wave = tid >> 5;
  const int lane = tid & 31;
  const int mrow = lane & 15;                 // A-frag row (M) per ISA layout
  const int kb8  = (lane >> 4) * 8;           // unused (A-frag K base is kb16)
  const int kb16 = (lane >> 4) * 16;          // B-frag K base offset
  const int ncol = lane & 15;                 // B/C/D-frag column (N)

  // Stage z tile into LDS as f16
  for (int i = tid; i < 16 * LDIM; i += 256) {
    int r = i >> 6, k = i & 63;
    zs[r * ZS_STRIDE + k] = (_Float16)z[(row0 + r) * LDIM + k];
  }
  __syncthreads();

  // ---- hidden = relu(z @ W1 + b1): 16 N-tiles of 16, 2 per wave ----
  for (int t = wave * 2; t < wave * 2 + 2; ++t) {
    const int col = t * 16 + ncol;
    v8f c = {};
    #pragma unroll
    for (int ks = 0; ks < LDIM / 32; ++ks) {
      // A-fragment: two aligned 16B LDS loads, contiguous K = 16h + s (matches B)
      v8h alo = *(const v8h*)&zs[mrow * ZS_STRIDE + ks * 32 + kb16];
      v8h ahi = *(const v8h*)&zs[mrow * ZS_STRIDE + ks * 32 + kb16 + 8];
      v16h a  = __builtin_shufflevector(alo, ahi,
                  0, 1, 2, 3, 4, 5, 6, 7, 8, 9, 10, 11, 12, 13, 14, 15);
      // B-fragment: one contiguous 32B load from transposed f16 weights
      v16h bf = *(const v16h*)&W1T[col * LDIM + ks * 32 + kb16];
      c = __builtin_amdgcn_wmma_f32_16x16x32_f16(
            false, a, false, bf, (short)0, c, false, false);
    }
    const float bias = b1[col];
    #pragma unroll
    for (int r = 0; r < 8; ++r) {
      const int M = r + ((lane >> 4) << 3);         // D-frag row per ISA layout
      float hv = fmaxf(c[r] + bias, 0.0f);
      hid[M * HID_STRIDE + col] = (_Float16)hv;
    }
  }
  __syncthreads();

  // ---- logits = hidden @ W2 + b2, then log_softmax over each C=16 group ----
  // A 16-column tile at colbase=16*t is exactly feature d=t's 16 categories.
  for (int t = wave * 4; t < wave * 4 + 4; ++t) {
    const int col = t * 16 + ncol;
    v8f c = {};
    #pragma unroll
    for (int ks = 0; ks < HDIM / 32; ++ks) {
      if (ks < HDIM / 32 - 1)   // speculative prefetch of next W2T K-slice
        __builtin_prefetch(&W2T[col * HDIM + (ks + 1) * 32 + kb16], 0, 3);
      v8h alo = *(const v8h*)&hid[mrow * HID_STRIDE + ks * 32 + kb16];
      v8h ahi = *(const v8h*)&hid[mrow * HID_STRIDE + ks * 32 + kb16 + 8];
      v16h a  = __builtin_shufflevector(alo, ahi,
                  0, 1, 2, 3, 4, 5, 6, 7, 8, 9, 10, 11, 12, 13, 14, 15);
      v16h bf = *(const v16h*)&W2T[col * HDIM + ks * 32 + kb16];
      c = __builtin_amdgcn_wmma_f32_16x16x32_f16(
            false, a, false, bf, (short)0, c, false, false);
    }
    const float bias = b2[col];
    #pragma unroll
    for (int r = 0; r < 8; ++r) {
      float v = c[r] + bias;
      // row-wise (over 16 categories == 16 lanes of a half-wave) max & sum
      float m = v;
      for (int msk = 1; msk <= 8; msk <<= 1)
        m = fmaxf(m, __shfl_xor(m, msk, 32));
      float e = __expf(v - m);
      float s = e;
      for (int msk = 1; msk <= 8; msk <<= 1)
        s += __shfl_xor(s, msk, 32);
      const float lp = (v - m) - __logf(s);
      const int M = r + ((lane >> 4) << 3);
      logp[(size_t)(row0 + M) * DC + col] = lp;
    }
  }
}

// -------------------------------------------------------------------------
// Kernel 2: scores[b,n] = sum_{d=D-YNUM..D-1} logp[n, d, x[b,d]]
// blockIdx.y = b  ->  x[] loads are scalar-uniform per block.
// -------------------------------------------------------------------------
__global__ __launch_bounds__(256) void scores_kernel(
    const float* __restrict__ logp, const int* __restrict__ x,
    float* __restrict__ scores)
{
  const int n = blockIdx.x * 256 + threadIdx.x;
  const int b = blockIdx.y;
  float s = 0.0f;
  #pragma unroll
  for (int j = 0; j < YNUM; ++j) {
    const int d = DFEAT - YNUM + j;
    const int c = x[b * DFEAT + d];
    s += logp[(size_t)n * DC + d * CCAT + c];
  }
  scores[(size_t)b * NBINS + n] = s;
}

// -------------------------------------------------------------------------
// Kernel 3: per batch row: deterministic top-16 (smaller index wins ties),
// num = full logp sum at selected bins, den = num - score, output = lse diff.
// -------------------------------------------------------------------------
__global__ __launch_bounds__(256) void topk_out_kernel(
    const float* __restrict__ scores, const float* __restrict__ logp,
    const int* __restrict__ x, float* __restrict__ out)
{
  __shared__ float s[NBINS];
  __shared__ float rv[256];
  __shared__ int   ri[256];
  __shared__ float tval[KTOP];
  __shared__ int   tidx[KTOP];
  __shared__ float numv[KTOP], denv[KTOP];

  const int b   = blockIdx.x;
  const int tid = threadIdx.x;

  for (int i = tid; i < NBINS; i += 256) s[i] = scores[(size_t)b * NBINS + i];
  __syncthreads();

  for (int round = 0; round < KTOP; ++round) {
    float best = -__builtin_inff();
    int   bi   = 0x7fffffff;
    for (int i = tid; i < NBINS; i += 256) {
      float v = s[i];
      if (v > best || (v == best && i < bi)) { best = v; bi = i; }
    }
    rv[tid] = best; ri[tid] = bi;
    __syncthreads();
    for (int off = 128; off > 0; off >>= 1) {
      if (tid < off) {
        float ov = rv[tid + off]; int oi = ri[tid + off];
        if (ov > rv[tid] || (ov == rv[tid] && oi < ri[tid])) {
          rv[tid] = ov; ri[tid] = oi;
        }
      }
      __syncthreads();
    }
    if (tid == 0) {
      tval[round] = rv[0];
      tidx[round] = ri[0];
      s[ri[0]] = -__builtin_inff();
    }
    __syncthreads();
  }

  if (tid < KTOP) {
    const int n = tidx[tid];
    float num = 0.0f;
    #pragma unroll
    for (int d = 0; d < DFEAT; ++d)
      num += logp[(size_t)n * DC + d * CCAT + x[b * DFEAT + d]];
    numv[tid] = num;
    denv[tid] = num - tval[tid];   // den drops the last YNUM (masked) features
  }
  __syncthreads();

  if (tid == 0) {
    float mn = -__builtin_inff(), md = -__builtin_inff();
    for (int j = 0; j < KTOP; ++j) {
      mn = fmaxf(mn, numv[j]); md = fmaxf(md, denv[j]);
    }
    float sn = 0.0f, sd = 0.0f;
    for (int j = 0; j < KTOP; ++j) {
      sn += __expf(numv[j] - mn); sd += __expf(denv[j] - md);
    }
    out[b] = (mn + __logf(sn)) - (md + __logf(sd));
  }
}

// -------------------------------------------------------------------------
extern "C" void kernel_launch(void* const* d_in, const int* in_sizes, int n_in,
                              void* d_out, int out_size, void* d_ws, size_t ws_size,
                              hipStream_t stream) {
  // setup_inputs order: x, log_w(unused), z, k(=16, hardcoded), W1, b1, W2, b2
  const int*   x  = (const int*)  d_in[0];
  const float* z  = (const float*)d_in[2];
  const float* W1 = (const float*)d_in[4];
  const float* b1 = (const float*)d_in[5];
  const float* W2 = (const float*)d_in[6];
  const float* b2 = (const float*)d_in[7];
  float* out = (float*)d_out;

  char* ws = (char*)d_ws;
  float*    logp   = (float*)ws;                                    // 16 MB
  float*    scores = (float*)(ws + (size_t)NBINS * DC * 4);         //  8 MB
  _Float16* W1T    = (_Float16*)(ws + (size_t)NBINS * DC * 4
                                    + (size_t)BATCH * NBINS * 4);   // 32 KB
  _Float16* W2T    = (_Float16*)((char*)W1T + (size_t)LDIM * HDIM * 2); // 256 KB

  const int prep_elems = LDIM * HDIM + HDIM * DC;
  prep_kernel<<<dim3((prep_elems + 255) / 256), dim3(256), 0, stream>>>(W1, W2, W1T, W2T);
  mlp_logp_kernel<<<dim3(NBINS / 16), dim3(256), 0, stream>>>(z, W1T, b1, W2T, b2, logp);
  scores_kernel<<<dim3(NBINS / 256, BATCH), dim3(256), 0, stream>>>(logp, x, scores);
  topk_out_kernel<<<dim3(BATCH), dim3(256), 0, stream>>>(scores, logp, x, out);
}
